// EpisodicMemory_41652592836910
// MI455X (gfx1250) — matrix operand order repaired and checked by reading, not compiled, not run on hardware
//
#include <hip/hip_runtime.h>
#include <hip/hip_bf16.h>

typedef __attribute__((ext_vector_type(16))) __bf16 v16bf;
typedef __attribute__((ext_vector_type(8)))  __bf16 v8bf;
typedef __attribute__((ext_vector_type(8)))  float  v8f;

#define BQ   4096
#define CAP  8192
#define DDIM 512
#define TOPK 64
#define BETA_SCALE 1.0f

// GEMM tiling
#define BM 128              // block tile rows
#define BN 64               // block tile cols
#define KSTEP 32            // K per stage
#define LDS_STRIDE 40       // padded row stride in bf16 elems (80B = 20 banks, 16B aligned)

__device__ __forceinline__ unsigned short f32_to_bf16_rne(float x) {
    unsigned int u = __float_as_uint(x);
    unsigned int r = u + 0x7FFFu + ((u >> 16) & 1u);
    return (unsigned short)(r >> 16);
}

// Per-lane async 16B copy: global -> LDS, tracked by ASYNCcnt.
// asm form (ISA 15.18.3): global_load_async_to_lds_b128 vdst(LDS addr), vaddr(64b), off
__device__ __forceinline__ void async_copy_b128(const unsigned short* g,
                                                const unsigned short* l) {
    unsigned lds_addr = (unsigned)(unsigned long long)
        (const __attribute__((address_space(3))) unsigned short*)l;
    unsigned long long gaddr = (unsigned long long)g;
    asm volatile("global_load_async_to_lds_b128 %0, %1, off"
                 :: "v"(lds_addr), "v"(gaddr)
                 : "memory");
}

__device__ __forceinline__ void async_wait_all() {
    asm volatile("s_wait_asynccnt 0x0" ::: "memory");
}

// Copy a (rows x 32) bf16 tile, global -> LDS (padded stride), 16B chunks.
__device__ __forceinline__ void copy_tile(const unsigned short* __restrict__ gsrc,
                                          int gRowStride,
                                          unsigned short* ldst,
                                          int rows, int tid) {
    const int nchunks = rows * 4;                 // 4 x 16B chunks per 32-elem row
    for (int c = tid; c < nchunks; c += 256) {
        const int row = c >> 2;
        const int kp  = (c & 3) * 8;              // elem offset within row
        async_copy_b128(gsrc + (size_t)row * gRowStride + kp,
                        ldst + row * LDS_STRIDE + kp);
    }
}

// ---------------------------------------------------------------- convert
__global__ void __launch_bounds__(256)
cvt_f32_to_bf16(const float* __restrict__ in, unsigned short* __restrict__ out, int n) {
    int i = blockIdx.x * blockDim.x + threadIdx.x;
    int stride = gridDim.x * blockDim.x;
    for (; i < n; i += stride) out[i] = f32_to_bf16_rne(in[i]);
}

// ---------------------------------------------------------------- WMMA GEMM
// C[M,N] = alpha * A[M,K] @ B[N,K]^T.  A,B bf16 row-major (ushort storage).
// Block = 256 threads = 8 wave32s in a 4(M) x 2(N) grid of 32x32 wave tiles
// (2x2 WMMA register tiles each) -> 128x64 block tile. K double-buffered
// through LDS via async global->LDS loads (ASYNCcnt).
template <int OUT_BF16>
__global__ void __launch_bounds__(256)
gemm_bt_wmma(const unsigned short* __restrict__ A,
             const unsigned short* __restrict__ B,
             float* __restrict__ Cf,
             unsigned short* __restrict__ Cb,
             int M, int N, int K, float alpha) {
    __shared__ unsigned short lA[2][BM * LDS_STRIDE];
    __shared__ unsigned short lB[2][BN * LDS_STRIDE];

    const int tid  = threadIdx.x;
    const int lane = tid & 31;
    const int wave = tid >> 5;
    const int wr   = wave & 3;              // wave row  (4)
    const int wc   = wave >> 2;             // wave col  (2)
    const int half = lane >> 4;             // 16-lane half
    const int l16  = lane & 15;

    const int tileM0 = blockIdx.y * BM;
    const int tileN0 = blockIdx.x * BN;

    // prologue: stage K-slab 0 into buffer 0
    copy_tile(A + (size_t)tileM0 * K, K, lA[0], BM, tid);
    copy_tile(B + (size_t)tileN0 * K, K, lB[0], BN, tid);

    v8f c[2][2] = {};
    int buf = 0;

    for (int kk = 0; kk < K; kk += KSTEP) {
        async_wait_all();
        __syncthreads();                    // buf ready; next buf free of readers

        const int nbuf = buf ^ 1;
        if (kk + KSTEP < K) {
            copy_tile(A + (size_t)tileM0 * K + kk + KSTEP, K, lA[nbuf], BM, tid);
            copy_tile(B + (size_t)tileN0 * K + kk + KSTEP, K, lB[nbuf], BN, tid);
        }

        // fragments from LDS
        const unsigned short* sA = lA[buf];
        const unsigned short* sB = lB[buf];
        v16bf a[2], b[2];
        #pragma unroll
        for (int i = 0; i < 2; ++i) {
            // A: lane half 0 -> K {0..15}, half 1 -> K {16..31}
            const unsigned short* pa = sA + (wr * 32 + i * 16 + l16) * LDS_STRIDE;
            v8bf x0 = *(const v8bf*)(pa + half * 16);
            v8bf x1 = *(const v8bf*)(pa + half * 16 + 8);
            #pragma unroll
            for (int e = 0; e < 8; ++e) { a[i][e] = x0[e]; a[i][8 + e] = x1[e]; }
            // B: lane l16 = column n; half selects K block of 16 (contiguous)
            const unsigned short* pb = sB + (wc * 32 + i * 16 + l16) * LDS_STRIDE;
            b[i] = *(const v16bf*)(pb + half * 16);
        }

        #pragma unroll
        for (int i = 0; i < 2; ++i)
            #pragma unroll
            for (int j = 0; j < 2; ++j)
                c[i][j] = __builtin_amdgcn_wmma_f32_16x16x32_bf16(
                              false, a[i], false, b[j],
                              (short)0, c[i][j], false, false);

        __syncthreads();                    // done reading buf before overwrite
        buf = nbuf;
    }

    // C/D layout: VGPR r -> row +half*8+r, col +l16
    #pragma unroll
    for (int i = 0; i < 2; ++i) {
        #pragma unroll
        for (int j = 0; j < 2; ++j) {
            #pragma unroll
            for (int r = 0; r < 8; ++r) {
                const size_t row = (size_t)(tileM0 + wr * 32 + i * 16 + half * 8 + r);
                const size_t col = (size_t)(tileN0 + wc * 32 + j * 16 + l16);
                const float  val = alpha * c[i][j][r];
                if (OUT_BF16) Cb[row * N + col] = f32_to_bf16_rne(val);
                else          Cf[row * N + col] = val;
            }
        }
    }
}

// ---------------------------------------------------------------- top-k + softmax + AV
// One block per query row. Scores row lives in LDS (32KB of the 320KB/WGP).
__global__ void __launch_bounds__(256)
topk_softmax_av(const float* __restrict__ scores,
                const float* __restrict__ v,
                float* __restrict__ out) {
    __shared__ float s[CAP];
    __shared__ float rv[256];
    __shared__ int   ri[256];
    __shared__ float topv[TOPK];
    __shared__ int   topi[TOPK];
    __shared__ float wexp[TOPK];
    __shared__ float wsum;

    const int row = blockIdx.x;
    const int tid = threadIdx.x;
    const float* srow = scores + (size_t)row * CAP;

    for (int i = tid; i < CAP; i += 256) s[i] = srow[i];
    __syncthreads();

    const float NEG_INF = -__builtin_inff();
    for (int sel = 0; sel < TOPK; ++sel) {
        float bv = NEG_INF; int bi = 0x7FFFFFFF;
        for (int i = tid; i < CAP; i += 256) {
            float x = s[i];
            if (x > bv) { bv = x; bi = i; }
        }
        rv[tid] = bv; ri[tid] = bi;
        __syncthreads();
        for (int off = 128; off > 0; off >>= 1) {
            if (tid < off) {
                float ov = rv[tid + off]; int oi = ri[tid + off];
                if (ov > rv[tid] || (ov == rv[tid] && oi < ri[tid])) {
                    rv[tid] = ov; ri[tid] = oi;
                }
            }
            __syncthreads();
        }
        if (tid == 0) {
            topv[sel] = rv[0];
            topi[sel] = ri[0];
            s[ri[0]]  = NEG_INF;   // remove winner
        }
        __syncthreads();
    }

    if (tid < TOPK) wexp[tid] = __expf(topv[tid] - topv[0]);
    __syncthreads();
    if (tid == 0) {
        float acc = 0.f;
        for (int j = 0; j < TOPK; ++j) acc += wexp[j];
        wsum = acc;
    }
    __syncthreads();

    const float inv = 1.0f / wsum;
    for (int d = tid; d < DDIM; d += 256) {
        float acc = 0.f;
        #pragma unroll 8
        for (int j = 0; j < TOPK; ++j)
            acc += wexp[j] * v[(size_t)topi[j] * DDIM + d];
        out[(size_t)row * DDIM + d] = acc * inv;
    }
}

// ---------------------------------------------------------------- launch
extern "C" void kernel_launch(void* const* d_in, const int* in_sizes, int n_in,
                              void* d_out, int out_size, void* d_ws, size_t ws_size,
                              hipStream_t stream) {
    const float* query  = (const float*)d_in[0];   // (BQ, D)
    const float* memory = (const float*)d_in[1];   // (CAP, D)
    const float* Wq     = (const float*)d_in[2];   // (D, D)
    const float* Wk     = (const float*)d_in[3];
    const float* Wv     = (const float*)d_in[4];
    // d_in[5] = top_k (device scalar); fixed at 64 per setup_inputs.

    char* ws = (char*)d_ws;
    size_t off = 0;
    auto alloc = [&](size_t bytes) -> char* {
        char* p = ws + off;
        off += (bytes + 255) & ~(size_t)255;
        return p;
    };

    unsigned short* usQ  = (unsigned short*)alloc((size_t)BQ  * DDIM * 2); // query bf16
    unsigned short* usM  = (unsigned short*)alloc((size_t)CAP * DDIM * 2); // memory bf16
    unsigned short* usWq = (unsigned short*)alloc((size_t)DDIM * DDIM * 2);
    unsigned short* usWk = (unsigned short*)alloc((size_t)DDIM * DDIM * 2);
    unsigned short* usWv = (unsigned short*)alloc((size_t)DDIM * DDIM * 2);
    unsigned short* usQp = (unsigned short*)alloc((size_t)BQ  * DDIM * 2); // q proj bf16
    unsigned short* usKp = (unsigned short*)alloc((size_t)CAP * DDIM * 2); // k proj bf16
    float*          fV   = (float*)alloc((size_t)CAP * DDIM * 4);          // v proj f32
    float*          fS   = (float*)alloc((size_t)BQ  * CAP * 4);           // scores f32

    // 1) f32 -> bf16 conversions
    {
        int n;
        n = BQ * DDIM;
        cvt_f32_to_bf16<<<(n + 255) / 256, 256, 0, stream>>>(query, usQ, n);
        n = CAP * DDIM;
        cvt_f32_to_bf16<<<(n + 255) / 256, 256, 0, stream>>>(memory, usM, n);
        n = DDIM * DDIM;
        cvt_f32_to_bf16<<<(n + 255) / 256, 256, 0, stream>>>(Wq, usWq, n);
        cvt_f32_to_bf16<<<(n + 255) / 256, 256, 0, stream>>>(Wk, usWk, n);
        cvt_f32_to_bf16<<<(n + 255) / 256, 256, 0, stream>>>(Wv, usWv, n);
    }

    // 2) projections: y = x @ W.T  (C = A * B^T with B = W as N x K)
    gemm_bt_wmma<1><<<dim3(DDIM / BN, BQ / BM), 256, 0, stream>>>(
        usQ, usWq, nullptr, usQp, BQ, DDIM, DDIM, 1.0f);
    gemm_bt_wmma<1><<<dim3(DDIM / BN, CAP / BM), 256, 0, stream>>>(
        usM, usWk, nullptr, usKp, CAP, DDIM, DDIM, 1.0f);
    gemm_bt_wmma<0><<<dim3(DDIM / BN, CAP / BM), 256, 0, stream>>>(
        usM, usWv, fV, nullptr, CAP, DDIM, DDIM, 1.0f);

    // 3) scores = BETA * q @ k.T  (4096 x 8192), stays L2-resident (128MB < 192MB)
    gemm_bt_wmma<0><<<dim3(CAP / BN, BQ / BM), 256, 0, stream>>>(
        usQp, usKp, fS, nullptr, BQ, CAP, DDIM, BETA_SCALE);

    // 4) per-row top-64 + softmax + weighted gather of v
    topk_softmax_av<<<BQ, 256, 0, stream>>>(fS, fV, (float*)d_out);
}
